// RetinaNetHead_2267742732678
// MI455X (gfx1250) — compile-verified, matrix operand order checked by reading, not run on hardware
//
#include <hip/hip_runtime.h>
#include <hip/hip_bf16.h>
#include <math.h>

typedef unsigned short u16t;
typedef unsigned int   u32t;
typedef __attribute__((ext_vector_type(16))) __bf16 v16bf;
typedef __attribute__((ext_vector_type(8)))  float  v8f;

#define CIN          256
#define KTOT         2304      // 9 * 256
#define ANUM         9
#define NCLS         80
#define MAXDET       100
#define CAND_CAP     5120
#define SCORE_THR_F  0.05f
#define NMS_THR_F    0.5f
#define CLASS_OFF_F  10000.0f
#define SCALE_CLAMP_F 4.135166556742356f  // log(1000/16)

static __device__ __forceinline__ u16t f32_to_bf16(float f) {
  u32t u = __float_as_uint(f);
  u32t r = u + 0x7FFFu + ((u >> 16) & 1u);     // round-to-nearest-even
  if ((u & 0x7F800000u) == 0x7F800000u) r = u; // inf/nan passthrough
  return (u16t)(r >> 16);
}

// ---------------------------------------------------------------------------
// Implicit-GEMM 3x3 conv, Cin=256, via V_WMMA_F32_16X16X32_BF16.
// Block = 256 threads = 8 waves. One 16-pixel spatial tile shared in LDS;
// each wave computes TWO adjacent 16-wide Cout slices (32 couts/wave,
// 256 couts per block) so the A fragment feeds 2 WMMAs per K-chunk.
// K = 9*256, walked as 9 fully-unrolled taps x 8 chunks of 32; tap coords,
// halo bounds and base pointers are compile-time-hoisted per tap.
// ---------------------------------------------------------------------------
template <bool OUT_F32>
__global__ __launch_bounds__(256) void conv3x3_wmma(
    const u16t* __restrict__ in,   // [H][W][256] bf16
    const u16t* __restrict__ wt,   // [coutPad][2304] bf16 (zero padded)
    const float* __restrict__ bias,
    void* __restrict__ out,        // [HW][coutPad] bf16 or f32
    int H, int W, int coutPad, int coutReal, int doRelu) {
  // rows padded to 40 shorts (80B): keeps b128 alignment, spreads banks
  __shared__ u16t As[16][40];
  __shared__ u16t Bs[256][40];

  const int tid  = threadIdx.x;
  const int wave = tid >> 5, lane = tid & 31;
  const int tile = blockIdx.x;
  const int coutBlk = blockIdx.y * 256;
  const int mn = lane & 15, half = lane >> 4;

  v8f acc0 = {}, acc1 = {};

  // A-staging coordinates for this thread (one u32 = 2 bf16 per chunk)
  const int rA  = tid >> 4;
  const int kkA = (tid & 15) << 1;
  const int pfA = tile * 16 + rA;
  const int yA  = pfA / W, xA = pfA % W;

  // B-staging: 4 x uint4 per thread per chunk; pointers hoisted out of loop
  const u16t* wptr[4];
  u16t*       bdst[4];
#pragma unroll
  for (int i = 0; i < 4; ++i) {
    const int q  = tid + (i << 8);            // 0..1023
    const int cl = q >> 2, ko = (q & 3) << 3; // cout-local row, K offset
    wptr[i] = wt + (size_t)(coutBlk + cl) * KTOT + ko;
    bdst[i] = &Bs[cl][ko];
  }

  const int brow = (wave << 5) + mn;          // this lane's B column rows

#pragma unroll
  for (int plane = 0; plane < 9; ++plane) {   // taps fully unrolled: ky/kx const
    const int ky = plane / 3 - 1, kx = plane % 3 - 1;
    const int yy = yA + ky, xx = xA + kx;
    const bool inb = (unsigned)yy < (unsigned)H && (unsigned)xx < (unsigned)W;
    // clamp to a safe address so the halo load is unconditional (cndmask later)
    const u16t* aBase = inb ? in + ((size_t)(yy * W + xx) * CIN + kkA)
                            : in + kkA;
#pragma unroll 1
    for (int cc = 0; cc < 8; ++cc) {
      const int kbase = (plane << 8) + (cc << 5);
      __syncthreads();
      // ---- stage A tile: 16 pixels x 32 channels, zero-padded halo ----
      const u32t raw = *(const u32t*)(aBase + (cc << 5));
      *(u32t*)&As[rA][kkA] = inb ? raw : 0u;
      // ---- stage B tile: 256 couts x 32 K, 16B vector loads ----
      __builtin_prefetch(wptr[0] + kbase + 32, 0, 0);  // global_prefetch_b8
#pragma unroll
      for (int i = 0; i < 4; ++i)
        *(uint4*)bdst[i] = *(const uint4*)(wptr[i] + kbase);
      __syncthreads();
      // ---- fragments per 16-bit WMMA VGPR layout ----
      // lane L: row/col = L%16; K halves {0..7,16..23} (L<16) / {8..15,24..31}
      union { uint4 q[2]; v16bf v; } fa, fb0, fb1;
      fa.q[0]  = *(const uint4*)&As[mn][half << 3];
      fa.q[1]  = *(const uint4*)&As[mn][16 + (half << 3)];
      fb0.q[0] = *(const uint4*)&Bs[brow][half << 3];
      fb0.q[1] = *(const uint4*)&Bs[brow][16 + (half << 3)];
      fb1.q[0] = *(const uint4*)&Bs[brow + 16][half << 3];
      fb1.q[1] = *(const uint4*)&Bs[brow + 16][16 + (half << 3)];
      acc0 = __builtin_amdgcn_wmma_f32_16x16x32_bf16(
          false, fa.v, false, fb0.v, (short)0, acc0, false, false);
      acc1 = __builtin_amdgcn_wmma_f32_16x16x32_bf16(
          false, fa.v, false, fb1.v, (short)0, acc1, false, false);
    }
  }

  // ---- epilogue: C/D layout lane L -> N=L%16, M=vgpr + 8*(L/16) ----
  union { v8f v; float f[8]; } r0, r1; r0.v = acc0; r1.v = acc1;
  const int cout0 = coutBlk + (wave << 5) + mn;
  const int cout1 = cout0 + 16;
  const float bv0 = (cout0 < coutReal) ? bias[cout0] : 0.f;
  const float bv1 = (cout1 < coutReal) ? bias[cout1] : 0.f;
#pragma unroll
  for (int i = 0; i < 8; ++i) {
    const int m = i + (half << 3);
    const size_t row = (size_t)(tile * 16 + m) * coutPad;
    float v0 = r0.f[i] + bv0;
    float v1 = r1.f[i] + bv1;
    if (doRelu) { v0 = fmaxf(v0, 0.f); v1 = fmaxf(v1, 0.f); }
    if (OUT_F32) {
      ((float*)out)[row + cout0] = v0;
      ((float*)out)[row + cout1] = v1;
    } else {
      ((u16t*)out)[row + cout0] = f32_to_bf16(v0);
      ((u16t*)out)[row + cout1] = f32_to_bf16(v1);
    }
  }
}

// ---------------------------------------------------------------------------
// Repack HWIO fp32 weights -> bf16 [coutPad][K], K = (tap*256 + cin)
// ---------------------------------------------------------------------------
__global__ void pack_weights(const float* __restrict__ w, u16t* __restrict__ out,
                             int coutReal, int coutPad) {
  const int gid = blockIdx.x * blockDim.x + threadIdx.x;
  if (gid >= coutPad * KTOT) return;
  const int co = gid / KTOT, k = gid - co * KTOT;
  float v = 0.f;
  if (co < coutReal) {
    const int plane = k >> 8, ci = k & 255;
    v = w[(size_t)(plane * 256 + ci) * coutReal + co];
  }
  out[gid] = f32_to_bf16(v);
}

__global__ void cvt_f32_bf16(const float* __restrict__ in, u16t* __restrict__ out, int n) {
  const int gid = blockIdx.x * blockDim.x + threadIdx.x;
  if (gid < n) out[gid] = f32_to_bf16(in[gid]);
}

__global__ void clear_kernel(float* __restrict__ out, int n, int* __restrict__ counter) {
  const int gid = blockIdx.x * blockDim.x + threadIdx.x;
  if (gid < n) out[gid] = 0.f;
  if (gid == 0) *counter = 0;
}

// ---------------------------------------------------------------------------
// sigmoid + threshold + anchor decode -> candidate list
// ---------------------------------------------------------------------------
__global__ void score_decode(const float* __restrict__ logits, const float* __restrict__ reg,
                             const float* __restrict__ anchors, float* __restrict__ cand,
                             int* __restrict__ counter, int HW, int clsStride, int regStride) {
  const int gid = blockIdx.x * blockDim.x + threadIdx.x;
  if (gid >= HW * ANUM * NCLS) return;
  const int pix = gid / (ANUM * NCLS);
  const int rem = gid - pix * (ANUM * NCLS);
  const int a = rem / NCLS, c = rem - a * NCLS;
  const float lg = logits[(size_t)pix * clsStride + a * NCLS + c];
  const float s = 1.f / (1.f + __expf(-lg));
  if (s <= SCORE_THR_F) return;
  const int pos = atomicAdd(counter, 1);
  if (pos >= CAND_CAP) return;
  const float* d  = reg + (size_t)pix * regStride + a * 4;
  const float* an = anchors + (size_t)(pix * ANUM + a) * 4;
  const float w = an[2] - an[0], h = an[3] - an[1];
  const float cx = an[0] + 0.5f * w, cy = an[1] + 0.5f * h;
  const float dw = fminf(d[2], SCALE_CLAMP_F), dh = fminf(d[3], SCALE_CLAMP_F);
  const float pcx = d[0] * w + cx, pcy = d[1] * h + cy;
  const float pw = __expf(dw) * w, ph = __expf(dh) * h;
  float* o = cand + (size_t)pos * 6;
  o[0] = pcx - 0.5f * pw; o[1] = pcy - 0.5f * ph;
  o[2] = pcx + 0.5f * pw; o[3] = pcy + 0.5f * ph;
  o[4] = s; o[5] = (float)c;
}

// ---------------------------------------------------------------------------
// Greedy class-offset NMS, single block. out = [boxes 100x4 | scores 100 | cls 100]
// ---------------------------------------------------------------------------
__global__ __launch_bounds__(256) void nms_kernel(const float* __restrict__ cand,
                                                  const int* __restrict__ cnt,
                                                  float* __restrict__ out) {
  __shared__ float sc[CAND_CAP];
  __shared__ float rb[256];
  __shared__ int   ri[256];
  const int tid = threadIdx.x;
  int n = *cnt; if (n > CAND_CAP) n = CAND_CAP;
  for (int i = tid; i < CAND_CAP; i += 256) sc[i] = (i < n) ? cand[(size_t)i * 6 + 4] : 0.f;
  __syncthreads();
  for (int det = 0; det < MAXDET; ++det) {
    float best = 0.f; int bi = -1;
    for (int i = tid; i < n; i += 256) { const float v = sc[i]; if (v > best) { best = v; bi = i; } }
    rb[tid] = best; ri[tid] = bi; __syncthreads();
    for (int s2 = 128; s2 > 0; s2 >>= 1) {
      if (tid < s2 && rb[tid + s2] > rb[tid]) { rb[tid] = rb[tid + s2]; ri[tid] = ri[tid + s2]; }
      __syncthreads();
    }
    best = rb[0]; bi = ri[0];
    if (tid == 0) {
      if (best > 0.f && bi >= 0) {
        out[det * 4 + 0] = cand[(size_t)bi * 6 + 0];
        out[det * 4 + 1] = cand[(size_t)bi * 6 + 1];
        out[det * 4 + 2] = cand[(size_t)bi * 6 + 2];
        out[det * 4 + 3] = cand[(size_t)bi * 6 + 3];
        out[MAXDET * 4 + det] = best;
        ((int*)out)[MAXDET * 5 + det] = (int)cand[(size_t)bi * 6 + 5];
      } else {
        out[det * 4 + 0] = out[det * 4 + 1] = out[det * 4 + 2] = out[det * 4 + 3] = 0.f;
        out[MAXDET * 4 + det] = 0.f;
        ((int*)out)[MAXDET * 5 + det] = -1;
      }
    }
    if (best > 0.f && bi >= 0) {
      const float offc = cand[(size_t)bi * 6 + 5] * CLASS_OFF_F;
      const float bx1 = cand[(size_t)bi * 6 + 0] + offc, by1 = cand[(size_t)bi * 6 + 1] + offc;
      const float bx2 = cand[(size_t)bi * 6 + 2] + offc, by2 = cand[(size_t)bi * 6 + 3] + offc;
      const float barea = (bx2 - bx1) * (by2 - by1);
      for (int i = tid; i < n; i += 256) {
        if (sc[i] <= 0.f) continue;
        const float o2 = cand[(size_t)i * 6 + 5] * CLASS_OFF_F;
        const float x1 = cand[(size_t)i * 6 + 0] + o2, y1 = cand[(size_t)i * 6 + 1] + o2;
        const float x2 = cand[(size_t)i * 6 + 2] + o2, y2 = cand[(size_t)i * 6 + 3] + o2;
        const float iw = fminf(bx2, x2) - fmaxf(bx1, x1);
        const float ih = fminf(by2, y2) - fmaxf(by1, y1);
        const float inter = fmaxf(iw, 0.f) * fmaxf(ih, 0.f);
        const float area = (x2 - x1) * (y2 - y1);
        const float iou = inter / (area + barea - inter + 1e-9f);
        if (iou > NMS_THR_F) sc[i] = 0.f;
      }
      if (tid == 0) sc[bi] = 0.f;
    }
    __syncthreads();
  }
}

// ---------------------------------------------------------------------------
extern "C" void kernel_launch(void* const* d_in, const int* in_sizes, int n_in,
                              void* d_out, int out_size, void* d_ws, size_t ws_size,
                              hipStream_t stream) {
  (void)in_sizes; (void)n_in; (void)ws_size;
  const float* feat[5];    for (int l = 0; l < 5; ++l) feat[l]    = (const float*)d_in[l];
  const float* cls_w = (const float*)d_in[5];  const float* cls_b = (const float*)d_in[6];
  const float* box_w = (const float*)d_in[7];  const float* box_b = (const float*)d_in[8];
  const float* sc_w  = (const float*)d_in[9];  const float* sc_b  = (const float*)d_in[10];
  const float* bb_w  = (const float*)d_in[11]; const float* bb_b  = (const float*)d_in[12];
  const float* anchors[5]; for (int l = 0; l < 5; ++l) anchors[l] = (const float*)d_in[13 + l];
  static const int HWs[5] = {64, 32, 16, 8, 4};

  char* base = (char*)d_ws;
  size_t off = 0;
  auto alloc = [&](size_t bytes) -> char* {
    char* p = base + off; off = (off + bytes + 255) & ~(size_t)255; return p;
  };
  u16t* wCls[4]; u16t* wBox[4];
  for (int i = 0; i < 4; ++i) wCls[i] = (u16t*)alloc((size_t)256 * KTOT * 2);
  for (int i = 0; i < 4; ++i) wBox[i] = (u16t*)alloc((size_t)256 * KTOT * 2);
  u16t* wScore = (u16t*)alloc((size_t)768 * KTOT * 2);
  u16t* wBbox  = (u16t*)alloc((size_t)256 * KTOT * 2);
  u16t* featbf = (u16t*)alloc((size_t)4096 * 256 * 2);
  u16t* ping   = (u16t*)alloc((size_t)4096 * 256 * 2);
  u16t* pong   = (u16t*)alloc((size_t)4096 * 256 * 2);
  float* logits = (float*)alloc((size_t)4096 * 768 * 4);
  float* reg    = (float*)alloc((size_t)4096 * 256 * 4);
  float* cand   = (float*)alloc((size_t)CAND_CAP * 6 * 4);
  int*   counter = (int*)alloc(256);

  clear_kernel<<<4, 256, 0, stream>>>((float*)d_out, out_size, counter);

  // weight repack fp32 -> packed bf16 GEMM layout
  for (int i = 0; i < 4; ++i) {
    pack_weights<<<(256 * KTOT + 255) / 256, 256, 0, stream>>>(
        cls_w + (size_t)i * 9 * 256 * 256, wCls[i], 256, 256);
    pack_weights<<<(256 * KTOT + 255) / 256, 256, 0, stream>>>(
        box_w + (size_t)i * 9 * 256 * 256, wBox[i], 256, 256);
  }
  pack_weights<<<(768 * KTOT + 255) / 256, 256, 0, stream>>>(sc_w, wScore, 720, 768);
  pack_weights<<<(256 * KTOT + 255) / 256, 256, 0, stream>>>(bb_w, wBbox, 36, 256);

  for (int l = 0; l < 5; ++l) {
    const int Hh = HWs[l], Ww = HWs[l], HW = Hh * Ww;
    cvt_f32_bf16<<<(HW * 256 + 255) / 256, 256, 0, stream>>>(feat[l], featbf, HW * 256);
    const dim3 g(HW / 16, 1);   // 256 couts per block
    // cls tower
    conv3x3_wmma<false><<<g, 256, 0, stream>>>(featbf, wCls[0], cls_b + 0 * 256, ping, Hh, Ww, 256, 256, 1);
    conv3x3_wmma<false><<<g, 256, 0, stream>>>(ping,   wCls[1], cls_b + 1 * 256, pong, Hh, Ww, 256, 256, 1);
    conv3x3_wmma<false><<<g, 256, 0, stream>>>(pong,   wCls[2], cls_b + 2 * 256, ping, Hh, Ww, 256, 256, 1);
    conv3x3_wmma<false><<<g, 256, 0, stream>>>(ping,   wCls[3], cls_b + 3 * 256, pong, Hh, Ww, 256, 256, 1);
    conv3x3_wmma<true><<<dim3(HW / 16, 3), 256, 0, stream>>>(pong, wScore, sc_b, logits, Hh, Ww, 768, 720, 0);
    // box tower
    conv3x3_wmma<false><<<g, 256, 0, stream>>>(featbf, wBox[0], box_b + 0 * 256, ping, Hh, Ww, 256, 256, 1);
    conv3x3_wmma<false><<<g, 256, 0, stream>>>(ping,   wBox[1], box_b + 1 * 256, pong, Hh, Ww, 256, 256, 1);
    conv3x3_wmma<false><<<g, 256, 0, stream>>>(pong,   wBox[2], box_b + 2 * 256, ping, Hh, Ww, 256, 256, 1);
    conv3x3_wmma<false><<<g, 256, 0, stream>>>(ping,   wBox[3], box_b + 3 * 256, pong, Hh, Ww, 256, 256, 1);
    conv3x3_wmma<true><<<g, 256, 0, stream>>>(pong, wBbox, bb_b, reg, Hh, Ww, 256, 36, 0);
    // sigmoid + threshold + decode -> candidates
    const int tot = HW * ANUM * NCLS;
    score_decode<<<(tot + 255) / 256, 256, 0, stream>>>(
        logits, reg, anchors[l], cand, counter, HW, 768, 256);
  }
  nms_kernel<<<1, 256, 0, stream>>>(cand, counter, (float*)d_out);
}
